// MMSingleStreamBlockAudio_31765578121404
// MI455X (gfx1250) — matrix operand order repaired and checked by reading, not compiled, not run on hardware
//
#include <hip/hip_runtime.h>
#include <cstdint>
#include <cstddef>

// ---------------- constants (match reference) ----------------
#define HH   24
#define DD   128
#define HID  3072
#define MLPW 12288
#define TXT  256
#define IMG  2048
#define AUD  128
#define LL   (IMG + TXT)          // 2304
#define N1   (3*HID + MLPW)       // 21504
#define QKVW (3*HID)              // 9216
#define A2W  (HID + MLPW)         // 15360

// ---------------- types ----------------
typedef __bf16  v16bf __attribute__((ext_vector_type(16)));
typedef float   v8f   __attribute__((ext_vector_type(8)));
typedef unsigned int u32x4 __attribute__((ext_vector_type(4)));
typedef float   f32x4 __attribute__((ext_vector_type(4)));
typedef int     i32x4 __attribute__((ext_vector_type(4)));

union FragBF { v16bf v; u32x4 q[2]; unsigned int u[8]; };
union FragF  { v8f   v; float f[8]; };
union F4     { f32x4 v; float f[4]; };
union U4     { u32x4 v; unsigned short s[8]; };

#define AS1 __attribute__((address_space(1)))
#define AS3 __attribute__((address_space(3)))
typedef AS1 i32x4* gvec_ptr;
typedef AS3 i32x4* lvec_ptr;

#if defined(__HIP_DEVICE_COMPILE__) && __has_builtin(__builtin_amdgcn_global_load_async_to_lds_b128)
#define HAVE_ASYNC_LDS 1
#else
#define HAVE_ASYNC_LDS 0
#endif

__device__ __forceinline__ void async_cp16(const void* g, void* l) {
#if HAVE_ASYNC_LDS
    __builtin_amdgcn_global_load_async_to_lds_b128((gvec_ptr)g, (lvec_ptr)l, 0, 0);
#else
    (void)g; (void)l;
#endif
}

__device__ __forceinline__ void wait_asynccnt0() {
#if defined(__HIP_DEVICE_COMPILE__)
#if __has_builtin(__builtin_amdgcn_s_wait_asynccnt)
    __builtin_amdgcn_s_wait_asynccnt(0);
#else
    asm volatile("s_wait_asynccnt 0" ::: "memory");
#endif
#endif
}

__device__ __forceinline__ unsigned short f2bf(float f) {
    // hardware f32->bf16 convert
    __bf16 h = (__bf16)f;
    return __builtin_bit_cast(unsigned short, h);
}
__device__ __forceinline__ float gelu_tanh(float x) {
    float x3 = x * x * x;
    return 0.5f * x * (1.0f + tanhf(0.7978845608028654f * (x + 0.044715f * x3)));
}
__device__ __forceinline__ v8f wmma_bf16(v16bf a, v16bf b, v8f c) {
    return __builtin_amdgcn_wmma_f32_16x16x32_bf16(false, a, false, b, (short)0, c, false, false);
}

// ---------------- kernel 1: mod = silu(vec) @ mod_w + mod_b ----------------
__global__ __launch_bounds__(256) void mod_kernel(const float* __restrict__ vec,
                                                  const float* __restrict__ mw,
                                                  const float* __restrict__ mb,
                                                  float* __restrict__ mod) {
    int j = blockIdx.x * 256 + threadIdx.x;
    if (j >= QKVW) return;
    float acc = 0.f;
    for (int i = 0; i < HID; ++i) {
        float vv = vec[i];
        float s = vv / (1.f + __expf(-vv));
        acc += s * mw[(size_t)i * QKVW + j];
    }
    mod[j] = acc + mb[j];
}

// ---------------- kernel 2: LN (+ optional modulate) -> bf16 ----------------
__global__ __launch_bounds__(256) void ln_mod_kernel(const float* __restrict__ x,
                                                     const float* __restrict__ mod,
                                                     unsigned short* __restrict__ out) {
    int t = blockIdx.x, tid = threadIdx.x;
    const float* row = x + (size_t)t * HID;
    __shared__ float red[256];
    float s = 0.f;
    for (int c = tid; c < HID; c += 256) s += row[c];
    red[tid] = s; __syncthreads();
    for (int o = 128; o > 0; o >>= 1) { if (tid < o) red[tid] += red[tid + o]; __syncthreads(); }
    float mu = red[0] * (1.f / HID); __syncthreads();
    float v = 0.f;
    for (int c = tid; c < HID; c += 256) { float d = row[c] - mu; v += d * d; }
    red[tid] = v; __syncthreads();
    for (int o = 128; o > 0; o >>= 1) { if (tid < o) red[tid] += red[tid + o]; __syncthreads(); }
    float rstd = rsqrtf(red[0] * (1.f / HID) + 1e-6f);
    const float* shift = mod;
    const float* scale = mod ? mod + HID : nullptr;
    for (int c = tid; c < HID; c += 256) {
        float val = (row[c] - mu) * rstd;
        if (mod) val = val * (1.f + scale[c]) + shift[c];
        out[(size_t)t * HID + c] = f2bf(val);
    }
}

// ---------------- kernel 3: bf16 WMMA GEMM ----------------
// C = A(MxK bf16) @ B(KxN fp32, converted in staging) + bias
// mode 0: col<split -> outF[row*split+col] (fp32); else bf16(gelu) -> outG[row*ldG+colOffG+(col-split)]
// mode 2: outR[row*N+col] = xres[row*N+col] + val*gate[col]
__global__ __launch_bounds__(256) void gemm_kernel(const unsigned short* __restrict__ A,
                                                   const float* __restrict__ B,
                                                   const float* __restrict__ bias,
                                                   int M, int N, int K, int mode,
                                                   float* __restrict__ outF, int split,
                                                   unsigned short* __restrict__ outG, int ldG, int colOffG,
                                                   const float* __restrict__ xres,
                                                   const float* __restrict__ gate,
                                                   float* __restrict__ outR) {
    __shared__ __align__(16) unsigned short lA[128 * 32];
    __shared__ __align__(16) unsigned short lBT[128 * 32];
    int tid = threadIdx.x, lane = tid & 31, wid = tid >> 5;
    int wm = wid & 3, wn = wid >> 2;
    int bn = blockIdx.x * 128, bm = blockIdx.y * 128;
    int hi = (lane >= 16) ? 1 : 0, l15 = lane & 15;

    FragF acc[2][4];
#pragma unroll
    for (int i = 0; i < 2; ++i)
#pragma unroll
        for (int j = 0; j < 4; ++j)
#pragma unroll
            for (int v = 0; v < 8; ++v) acc[i][j].f[v] = 0.f;

    for (int k0 = 0; k0 < K; k0 += 32) {
        // stage A tile (128x32 bf16): CDNA5 async global->LDS copy if available
        {
            int r = tid >> 1, hseg = (tid & 1) * 16;
            const unsigned short* gp = A + (size_t)(bm + r) * K + k0 + hseg;
#if HAVE_ASYNC_LDS
            async_cp16(gp, &lA[r * 32 + hseg]);
            async_cp16(gp + 8, &lA[r * 32 + hseg + 8]);
#else
            const u32x4* g = (const u32x4*)gp;
            u32x4 d0 = g[0], d1 = g[1];
            *(u32x4*)&lA[r * 32 + hseg] = d0;
            *(u32x4*)&lA[r * 32 + hseg + 8] = d1;
#endif
        }
        // stage B tile (32x128 fp32) -> convert bf16 -> transposed BT[n][k]
#pragma unroll
        for (int j = 0; j < 4; ++j) {
            int idx = tid + j * 256;
            int kr = idx >> 5, seg = idx & 31;
            F4 w4; w4.v = *(const f32x4*)(B + (size_t)(k0 + kr) * N + bn + seg * 4);
            int nn = seg * 4;
#pragma unroll
            for (int e = 0; e < 4; ++e) lBT[(nn + e) * 32 + kr] = f2bf(w4.f[e]);
        }
        if (k0 + 32 < K) __builtin_prefetch(B + (size_t)(k0 + 32) * N + bn + (tid & 31) * 4, 0, 1);
#if HAVE_ASYNC_LDS
        wait_asynccnt0();
#endif
        __syncthreads();

        FragBF af[2], bfr[4];
#pragma unroll
        for (int i = 0; i < 2; ++i) {
            int m = wm * 32 + i * 16 + l15;
            int kk = hi ? 8 : 0;
            af[i].q[0] = *(const u32x4*)&lA[m * 32 + kk];
            af[i].q[1] = *(const u32x4*)&lA[m * 32 + kk + 16];
        }
#pragma unroll
        for (int j = 0; j < 4; ++j) {
            int n = wn * 64 + j * 16 + l15;
            int kk = hi ? 8 : 0;
            bfr[j].q[0] = *(const u32x4*)&lBT[n * 32 + kk];
            bfr[j].q[1] = *(const u32x4*)&lBT[n * 32 + kk + 16];
        }
#pragma unroll
        for (int i = 0; i < 2; ++i)
#pragma unroll
            for (int j = 0; j < 4; ++j)
                acc[i][j].v = wmma_bf16(af[i].v, bfr[j].v, acc[i][j].v);
        __syncthreads();
    }

    // epilogue
#pragma unroll
    for (int i = 0; i < 2; ++i)
#pragma unroll
        for (int j = 0; j < 4; ++j) {
            int col = bn + wn * 64 + j * 16 + l15;
            float bv = bias[col];
#pragma unroll
            for (int v = 0; v < 8; ++v) {
                int row = bm + wm * 32 + i * 16 + v + 8 * hi;
                float val = acc[i][j].f[v] + bv;
                if (mode == 0) {
                    if (col < split) outF[(size_t)row * split + col] = val;
                    else outG[(size_t)row * ldG + colOffG + (col - split)] = f2bf(gelu_tanh(val));
                } else {
                    outR[(size_t)row * N + col] = xres[(size_t)row * N + col] + val * gate[col];
                }
            }
        }
}

// ---------------- kernel 4: qkv post: RMS norm + RoPE -> bf16 q/k/v ----------------
__global__ __launch_bounds__(128) void qkv_post_kernel(const float* __restrict__ yqkv,
                                                       const float* __restrict__ qw,
                                                       const float* __restrict__ kw,
                                                       const float* __restrict__ fcos,
                                                       const float* __restrict__ fsin,
                                                       unsigned short* __restrict__ qb,
                                                       unsigned short* __restrict__ kb,
                                                       unsigned short* __restrict__ vb,
                                                       int imgLen) {
    int t = blockIdx.x / HH, h = blockIdx.x % HH, d = threadIdx.x;
    float q = yqkv[(((size_t)t * 3 + 0) * HH + h) * DD + d];
    float k = yqkv[(((size_t)t * 3 + 1) * HH + h) * DD + d];
    float v = yqkv[(((size_t)t * 3 + 2) * HH + h) * DD + d];
    __shared__ float red[128], qs[128], ks[128];
    red[d] = q * q; __syncthreads();
    for (int o = 64; o > 0; o >>= 1) { if (d < o) red[d] += red[d + o]; __syncthreads(); }
    float rq = rsqrtf(red[0] * (1.f / DD) + 1e-6f); __syncthreads();
    red[d] = k * k; __syncthreads();
    for (int o = 64; o > 0; o >>= 1) { if (d < o) red[d] += red[d + o]; __syncthreads(); }
    float rk = rsqrtf(red[0] * (1.f / DD) + 1e-6f); __syncthreads();
    float qn = q * rq * qw[d];
    float kn = k * rk * kw[d];
    if (t < imgLen) {
        qs[d] = qn; ks[d] = kn; __syncthreads();
        int p = d >> 1;
        float c = fcos[(size_t)t * (DD / 2) + p], s = fsin[(size_t)t * (DD / 2) + p];
        float qe = qs[p * 2], qo = qs[p * 2 + 1], ke = ks[p * 2], ko = ks[p * 2 + 1];
        qn = (d & 1) ? (qe * s + qo * c) : (qe * c - qo * s);
        kn = (d & 1) ? (ke * s + ko * c) : (ke * c - ko * s);
    }
    size_t o = ((size_t)t * HH + h) * DD + d;
    qb[o] = f2bf(qn); kb[o] = f2bf(kn); vb[o] = f2bf(v);
}

// ---------------- kernel 5: audio k/v post ----------------
__global__ __launch_bounds__(128) void aud_post_kernel(const float* __restrict__ aqkv,
                                                       const float* __restrict__ kw,
                                                       unsigned short* __restrict__ akb,
                                                       unsigned short* __restrict__ avb) {
    int t = blockIdx.x / HH, h = blockIdx.x % HH, d = threadIdx.x;
    float k = aqkv[(((size_t)t * 3 + 1) * HH + h) * DD + d];
    float v = aqkv[(((size_t)t * 3 + 2) * HH + h) * DD + d];
    __shared__ float red[128];
    red[d] = k * k; __syncthreads();
    for (int o = 64; o > 0; o >>= 1) { if (d < o) red[d] += red[d + o]; __syncthreads(); }
    float rk = rsqrtf(red[0] * (1.f / DD) + 1e-6f);
    size_t o = ((size_t)t * HH + h) * DD + d;
    akb[o] = f2bf(k * rk * kw[d]);
    avb[o] = f2bf(v);
}

// ---------------- kernel 6: flash attention (wave-level, WMMA) ----------------
__global__ __launch_bounds__(128) void attn_kernel(const unsigned short* __restrict__ Q,
                                                   const unsigned short* __restrict__ Kb,
                                                   const unsigned short* __restrict__ Vb,
                                                   int Lk, int nQT, int imgLen,
                                                   const int* __restrict__ txtMask,
                                                   float* __restrict__ outAttn,
                                                   int addMode, const float* __restrict__ scalePtr) {
    __shared__ __align__(16) unsigned short lP[4][16 * 32];
    __shared__ __align__(16) unsigned short lVT[4][128 * 32];
    int lane = threadIdx.x & 31, w = threadIdx.x >> 5;
    int gw = blockIdx.x * 4 + w;
    int h = gw / nQT, qt = gw % nQT;
    int hi = (lane >= 16) ? 1 : 0, l15 = lane & 15;
    unsigned short* P = lP[w];
    unsigned short* VT = lVT[w];

    // Q fragments (A layout), d split into 4 chunks of 32
    FragBF qa[4];
    {
        int tok = qt * 16 + (hi ? lane - 16 : lane);
        const unsigned short* qrow = Q + ((size_t)tok * HH + h) * DD;
#pragma unroll
        for (int c = 0; c < 4; ++c) {
            int kk = c * 32 + (hi ? 8 : 0);
            qa[c].q[0] = *(const u32x4*)(qrow + kk);
            qa[c].q[1] = *(const u32x4*)(qrow + kk + 16);
        }
    }

    FragF o[8];
    float mrow[8], lrow[8];
#pragma unroll
    for (int j = 0; j < 8; ++j) { mrow[j] = -1e30f; lrow[j] = 0.f; for (int v = 0; v < 8; ++v) o[j].f[v] = 0.f; }
    const float sscale = 0.08838834764831845f;  // 1/sqrt(128)

    for (int kt = 0; kt < Lk; kt += 32) {
        FragF s0, s1;
#pragma unroll
        for (int v = 0; v < 8; ++v) { s0.f[v] = 0.f; s1.f[v] = 0.f; }
        // S = Q @ K^T  (two 16-key tiles)
        {
            int key0 = kt + (hi ? lane - 16 : lane);
            const unsigned short* kr0 = Kb + ((size_t)key0 * HH + h) * DD;
            const unsigned short* kr1 = kr0 + (size_t)16 * HH * DD;
#pragma unroll
            for (int c = 0; c < 4; ++c) {
                int kk = c * 32 + (hi ? 8 : 0);
                FragBF b0, b1;
                b0.q[0] = *(const u32x4*)(kr0 + kk); b0.q[1] = *(const u32x4*)(kr0 + kk + 16);
                b1.q[0] = *(const u32x4*)(kr1 + kk); b1.q[1] = *(const u32x4*)(kr1 + kk + 16);
                s0.v = wmma_bf16(qa[c].v, b0.v, s0.v);
                s1.v = wmma_bf16(qa[c].v, b1.v, s1.v);
            }
        }
        // scale + mask (per-lane column)
        float add0 = 0.f, add1 = 0.f;
        if (txtMask) {
            int key0 = kt + l15, key1 = key0 + 16;
            if (key0 >= imgLen && txtMask[key0 - imgLen] <= 0) add0 = -1e9f;
            if (key1 >= imgLen && txtMask[key1 - imgLen] <= 0) add1 = -1e9f;
        }
        // online softmax per row
#pragma unroll
        for (int v = 0; v < 8; ++v) {
            float a0 = s0.f[v] * sscale + add0;
            float a1 = s1.f[v] * sscale + add1;
            float mx = fmaxf(a0, a1);
            mx = fmaxf(mx, __shfl_xor(mx, 1, 32));
            mx = fmaxf(mx, __shfl_xor(mx, 2, 32));
            mx = fmaxf(mx, __shfl_xor(mx, 4, 32));
            mx = fmaxf(mx, __shfl_xor(mx, 8, 32));
            float mn = fmaxf(mrow[v], mx);
            float alpha = __expf(mrow[v] - mn);
            mrow[v] = mn;
            float p0 = __expf(a0 - mn), p1 = __expf(a1 - mn);
            float rs = p0 + p1;
            rs += __shfl_xor(rs, 1, 32);
            rs += __shfl_xor(rs, 2, 32);
            rs += __shfl_xor(rs, 4, 32);
            rs += __shfl_xor(rs, 8, 32);
            lrow[v] = lrow[v] * alpha + rs;
#pragma unroll
            for (int j = 0; j < 8; ++j) o[j].f[v] *= alpha;
            P[(v + 8 * hi) * 32 + l15] = f2bf(p0);
            P[(v + 8 * hi) * 32 + 16 + l15] = f2bf(p1);
        }
        // stage V tile transposed: VT[d][key] (128x32)
        {
            int key = kt + lane;
            const unsigned short* vr = Vb + ((size_t)key * HH + h) * DD;
#pragma unroll
            for (int c = 0; c < 16; ++c) {
                U4 uu; uu.v = *(const u32x4*)(vr + c * 8);
#pragma unroll
                for (int e = 0; e < 8; ++e) VT[(c * 8 + e) * 32 + lane] = uu.s[e];
            }
        }
        __syncthreads();
        // P as A-frag (16x32), V^T as B-frags; O += P @ V
        FragBF pa;
        {
            int r = hi ? lane - 16 : lane;
            int kk = hi ? 8 : 0;
            pa.q[0] = *(const u32x4*)&P[r * 32 + kk];
            pa.q[1] = *(const u32x4*)&P[r * 32 + kk + 16];
        }
#pragma unroll
        for (int j = 0; j < 8; ++j) {
            int n = j * 16 + l15;
            int kk = hi ? 8 : 0;
            FragBF vbf;
            vbf.q[0] = *(const u32x4*)&VT[n * 32 + kk];
            vbf.q[1] = *(const u32x4*)&VT[n * 32 + kk + 16];
            o[j].v = wmma_bf16(pa.v, vbf.v, o[j].v);
        }
        __syncthreads();
    }
    // write / accumulate output (L, H*D) fp32
    float sc = 1.0f;
    if (addMode) sc = fminf(scalePtr[0], 2.0f);
#pragma unroll
    for (int j = 0; j < 8; ++j)
#pragma unroll
        for (int v = 0; v < 8; ++v) {
            int tok = qt * 16 + v + 8 * hi;
            int d = j * 16 + l15;
            float val = o[j].f[v] / lrow[v];
            size_t idx = (size_t)tok * (HH * DD) + h * DD + d;
            if (addMode) outAttn[idx] += val * sc;
            else outAttn[idx] = val;
        }
}

// ---------------- kernel 7: pack attn fp32 -> bf16 into a2[:, 0:3072] ----------------
__global__ __launch_bounds__(256) void pack_attn_kernel(const float* __restrict__ attn,
                                                        unsigned short* __restrict__ a2) {
    size_t i = (size_t)blockIdx.x * 256 + threadIdx.x;
    size_t t = i / HID, c = i % HID;
    a2[t * A2W + c] = f2bf(attn[i]);
}

// ---------------- host launcher ----------------
extern "C" void kernel_launch(void* const* d_in, const int* in_sizes, int n_in,
                              void* d_out, int out_size, void* d_ws, size_t ws_size,
                              hipStream_t stream) {
    const float* x          = (const float*)d_in[0];
    const float* vec        = (const float*)d_in[1];
    const float* fcos       = (const float*)d_in[2];
    const float* fsin       = (const float*)d_in[3];
    const int*   text_mask  = (const int*)d_in[4];
    const float* audio_emb  = (const float*)d_in[5];
    const float* w1         = (const float*)d_in[6];
    const float* b1         = (const float*)d_in[7];
    const float* w2         = (const float*)d_in[8];
    const float* b2         = (const float*)d_in[9];
    const float* q_norm_w   = (const float*)d_in[10];
    const float* k_norm_w   = (const float*)d_in[11];
    const float* mod_w      = (const float*)d_in[12];
    const float* mod_b      = (const float*)d_in[13];
    const float* aud_qkv_w  = (const float*)d_in[14];
    const float* aud_qkv_b  = (const float*)d_in[15];
    const float* aud_k_norm = (const float*)d_in[17];
    const float* aud_scale  = (const float*)d_in[18];
    float* out = (float*)d_out;

    size_t off = 0;
    auto alloc = [&](size_t bytes) -> void* {
        void* p = (char*)d_ws + off;
        off += (bytes + 255) & ~(size_t)255;
        return p;
    };
    float*          mod      = (float*)alloc((size_t)QKVW * 4);
    unsigned short* xmod     = (unsigned short*)alloc((size_t)LL * HID * 2);
    float*          yqkv     = (float*)alloc((size_t)LL * QKVW * 4);
    unsigned short* a2       = (unsigned short*)alloc((size_t)LL * A2W * 2);
    unsigned short* qb       = (unsigned short*)alloc((size_t)LL * HID * 2);
    unsigned short* kb       = (unsigned short*)alloc((size_t)LL * HID * 2);
    unsigned short* vb       = (unsigned short*)alloc((size_t)LL * HID * 2);
    float*          attn_f32 = (float*)alloc((size_t)LL * HID * 4);
    unsigned short* aud_ln   = (unsigned short*)alloc((size_t)AUD * HID * 2);
    float*          aud_qkv  = (float*)alloc((size_t)AUD * QKVW * 4);
    unsigned short* akb      = (unsigned short*)alloc((size_t)AUD * HID * 2);
    unsigned short* avb      = (unsigned short*)alloc((size_t)AUD * HID * 2);

    mod_kernel<<<QKVW / 256, 256, 0, stream>>>(vec, mod_w, mod_b, mod);
    ln_mod_kernel<<<LL, 256, 0, stream>>>(x, mod, xmod);
    ln_mod_kernel<<<AUD, 256, 0, stream>>>(audio_emb, nullptr, aud_ln);
    gemm_kernel<<<dim3(N1 / 128, LL / 128), 256, 0, stream>>>(
        xmod, w1, b1, LL, N1, HID, 0,
        yqkv, QKVW, a2, A2W, HID, nullptr, nullptr, nullptr);
    gemm_kernel<<<dim3(QKVW / 128, AUD / 128), 256, 0, stream>>>(
        aud_ln, aud_qkv_w, aud_qkv_b, AUD, QKVW, HID, 0,
        aud_qkv, QKVW, nullptr, 0, 0, nullptr, nullptr, nullptr);
    qkv_post_kernel<<<LL * HH, 128, 0, stream>>>(yqkv, q_norm_w, k_norm_w, fcos, fsin,
                                                 qb, kb, vb, IMG);
    aud_post_kernel<<<AUD * HH, 128, 0, stream>>>(aud_qkv, aud_k_norm, akb, avb);
    attn_kernel<<<(HH * (LL / 16)) / 4, 128, 0, stream>>>(
        qb, kb, vb, LL, LL / 16, IMG, text_mask, attn_f32, 0, nullptr);
    attn_kernel<<<(HH * (IMG / 16)) / 4, 128, 0, stream>>>(
        qb, akb, avb, AUD, IMG / 16, 1 << 30, nullptr, attn_f32, 1, aud_scale);
    pack_attn_kernel<<<(LL * HID) / 256, 256, 0, stream>>>(attn_f32, a2);
    gemm_kernel<<<dim3(HID / 128, LL / 128), 256, 0, stream>>>(
        a2, w2, b2, LL, HID, A2W, 2,
        nullptr, 0, nullptr, 0, 0, x, mod + 2 * HID, out);
    (void)in_sizes; (void)n_in; (void)out_size; (void)ws_size;
}
